// LigerLMHeadDPO_71975061946952
// MI455X (gfx1250) — matrix operand order repaired and checked by reading, not compiled, not run on hardware
//
#include <hip/hip_runtime.h>
#include <hip/hip_bf16.h>

typedef __attribute__((ext_vector_type(16))) __bf16 v16bf;
typedef __attribute__((ext_vector_type(8)))  float  v8f;

#define HDIM   4096
#define TOKS   4096          // B*T = 8*512
#define VOCAB  32000
#define NTILE  256
#define MTILE  128
#define KTILE  32
#define NT     (VOCAB / NTILE)   // 125 vocab chunks
#define KT     (HDIM / KTILE)    // 128 k-steps
#define SAP    40                // LDS row stride (halves), padded vs 32
#define SBP    40
#define IGNORE_IDX (-100)
#define BETA   0.1f

// f32x2 -> packed bf16x2 via the CDNA5 cvt instruction (clang builtin is not
// declared on this toolchain, but the assembler knows the gfx1250 mnemonic).
__device__ __forceinline__ unsigned bf16pk(float a, float b) {
    unsigned r;
    asm("v_cvt_pk_bf16_f32 %0, %1, %2" : "=v"(r) : "v"(a), "v"(b));
    return r;
}

union Frag { v16bf bf; float4 q[2]; };

// ---------------------------------------------------------------------------
// Kernel 1: bf16 WMMA GEMM tile + streaming logsumexp partials per vocab chunk
// grid = (NT, 2 models, 2 m-halves), block = 256 (8 wave32)
// Double-buffered LDS + register prefetch: 1 barrier per K-step, VMEM latency
// of tile k+1 hidden behind the 16 WMMAs of tile k.
// ---------------------------------------------------------------------------
__global__ __launch_bounds__(256)
void dpo_gemm_lse(const float* __restrict__ x, const float* __restrict__ rx,
                  const float* __restrict__ W, const float* __restrict__ rW,
                  float* __restrict__ partials) {
    __shared__ __align__(16) __bf16 sA[2 * MTILE * SAP];   // 20 KB
    __shared__ __align__(16) __bf16 sB[2 * NTILE * SBP];   // 40 KB
    __shared__ float redM[MTILE * 4];
    __shared__ float redS[MTILE * 4];

    const int tid  = threadIdx.x;
    const int lane = tid & 31;
    const int wave = tid >> 5;
    const int wm   = wave >> 2;   // 0..1 : 64-row group
    const int wn   = wave & 3;    // 0..3 : 64-col group
    const int lr   = lane & 15;
    const int lh   = lane >> 4;   // lane half

    const int model = blockIdx.y;
    const float* X  = model ? rx : x;
    const float* Wp = model ? rW : W;
    const int v0    = blockIdx.x * NTILE;
    const int mt0   = blockIdx.z * 16;

    // staging assignments (fixed per thread)
    const int arow = tid >> 1;            // A row 0..127
    const int ak   = (tid & 1) * 16;      // A k-half
    const float* gBbase = Wp + (size_t)(v0 + tid) * HDIM;   // B row = tid

    const v8f vzero = {0.f, 0.f, 0.f, 0.f, 0.f, 0.f, 0.f, 0.f};

    for (int mt = mt0; mt < mt0 + 16; ++mt) {
        const int tok0 = mt * MTILE;
        const float* gAbase = X + (size_t)(tok0 + arow) * HDIM + ak;

        v8f acc[4][4];
        #pragma unroll
        for (int mi = 0; mi < 4; ++mi)
            #pragma unroll
            for (int ni = 0; ni < 4; ++ni)
                acc[mi][ni] = vzero;

        float4 ga[4], gb[8];
        // ---- prologue: fetch K-tile 0 and stage into buffer 0 ----
        #pragma unroll
        for (int i = 0; i < 4; ++i)
            ga[i] = *reinterpret_cast<const float4*>(gAbase + 4 * i);
        #pragma unroll
        for (int i = 0; i < 8; ++i)
            gb[i] = *reinterpret_cast<const float4*>(gBbase + 4 * i);
        {
            __bf16* dA = &sA[arow * SAP + ak];
            #pragma unroll
            for (int i = 0; i < 4; ++i) {
                uint2 pk = { bf16pk(ga[i].x, ga[i].y), bf16pk(ga[i].z, ga[i].w) };
                *reinterpret_cast<uint2*>(dA + 4 * i) = pk;
            }
            __bf16* dB = &sB[tid * SBP];
            #pragma unroll
            for (int i = 0; i < 8; ++i) {
                uint2 pk = { bf16pk(gb[i].x, gb[i].y), bf16pk(gb[i].z, gb[i].w) };
                *reinterpret_cast<uint2*>(dB + 4 * i) = pk;
            }
        }

        for (int kt = 0; kt < KT; ++kt) {
            __syncthreads();    // buf[kt&1] visible; prior reads of buf[(kt+1)&1] done
            const int cur = (kt & 1);
            const int nxt = cur ^ 1;
            const bool more = (kt + 1) < KT;

            // ---- issue global prefetch of K-tile kt+1 (latency overlapped) ----
            if (more) {
                const int k1 = (kt + 1) * KTILE;
                #pragma unroll
                for (int i = 0; i < 4; ++i)
                    ga[i] = *reinterpret_cast<const float4*>(gAbase + k1 + 4 * i);
                #pragma unroll
                for (int i = 0; i < 8; ++i)
                    gb[i] = *reinterpret_cast<const float4*>(gBbase + k1 + 4 * i);
            }

            // ---- fragments from current LDS buffer ----
            const __bf16* bufA = &sA[cur * MTILE * SAP];
            const __bf16* bufB = &sB[cur * NTILE * SBP];
            Frag aF[4], bF[4];
            #pragma unroll
            for (int mi = 0; mi < 4; ++mi) {
                // A 16x32 bf16: lane holds row M=lane%16; lane-halves interleave K in 8-blocks
                const __bf16* p = bufA + (wm * 64 + mi * 16 + lr) * SAP + lh * 8;
                aF[mi].q[0] = *reinterpret_cast<const float4*>(p);
                aF[mi].q[1] = *reinterpret_cast<const float4*>(p + 16);
            }
            #pragma unroll
            for (int ni = 0; ni < 4; ++ni) {
                // B 32x16 bf16: lane holds col N=lane%16; lane-halves interleave K in 8-blocks
                const __bf16* p = bufB + (wn * 64 + ni * 16 + lr) * SBP + lh * 8;
                bF[ni].q[0] = *reinterpret_cast<const float4*>(p);
                bF[ni].q[1] = *reinterpret_cast<const float4*>(p + 16);
            }

            // ---- 16 WMMAs: 64x64 per-wave tile ----
            #pragma unroll
            for (int mi = 0; mi < 4; ++mi)
                #pragma unroll
                for (int ni = 0; ni < 4; ++ni)
                    acc[mi][ni] = __builtin_amdgcn_wmma_f32_16x16x32_bf16(
                        false, aF[mi].bf, false, bF[ni].bf,
                        (short)0, acc[mi][ni], false, false);

            // ---- convert + store prefetched tile into the other buffer ----
            if (more) {
                __bf16* dA = &sA[nxt * MTILE * SAP + arow * SAP + ak];
                #pragma unroll
                for (int i = 0; i < 4; ++i) {
                    uint2 pk = { bf16pk(ga[i].x, ga[i].y), bf16pk(ga[i].z, ga[i].w) };
                    *reinterpret_cast<uint2*>(dA + 4 * i) = pk;
                }
                __bf16* dB = &sB[nxt * NTILE * SBP + tid * SBP];
                #pragma unroll
                for (int i = 0; i < 8; ++i) {
                    uint2 pk = { bf16pk(gb[i].x, gb[i].y), bf16pk(gb[i].z, gb[i].w) };
                    *reinterpret_cast<uint2*>(dB + 4 * i) = pk;
                }
            }
        }

        // ---- per-row max / sum-exp over this 128x256 logits tile ----
        // C layout: VGPR r -> row (mi*16 + r + lh*8), lanes%16 -> col
        #pragma unroll
        for (int mi = 0; mi < 4; ++mi) {
            #pragma unroll
            for (int r = 0; r < 8; ++r) {
                float m = acc[mi][0][r];
                #pragma unroll
                for (int ni = 1; ni < 4; ++ni) m = fmaxf(m, acc[mi][ni][r]);
                #pragma unroll
                for (int off = 8; off >= 1; off >>= 1)
                    m = fmaxf(m, __shfl_xor(m, off, 32));   // stays in 16-lane half
                float s = 0.f;
                #pragma unroll
                for (int ni = 0; ni < 4; ++ni) s += __expf(acc[mi][ni][r] - m);
                #pragma unroll
                for (int off = 8; off >= 1; off >>= 1)
                    s += __shfl_xor(s, off, 32);
                const int row = wm * 64 + mi * 16 + r + lh * 8;
                if (lr == 0) { redM[row * 4 + wn] = m; redS[row * 4 + wn] = s; }
            }
        }
        __syncthreads();

        if (tid < MTILE) {
            const int row = tid;
            float M = redM[row * 4];
            #pragma unroll
            for (int j = 1; j < 4; ++j) M = fmaxf(M, redM[row * 4 + j]);
            float S = 0.f;
            #pragma unroll
            for (int j = 0; j < 4; ++j)
                S += redS[row * 4 + j] * __expf(redM[row * 4 + j] - M);
            const int token = tok0 + row;
            // layout [model][ntile][token][2] -> coalesced stores and reads
            float* pp = partials
                      + (((size_t)model * NT + blockIdx.x) * TOKS + token) * 2;
            pp[0] = M;
            pp[1] = S;
        }
        __syncthreads();   // red* reads done before next m-tile reuses them
    }
}

// ---------------------------------------------------------------------------
// Kernel 2: target logit = dot(x[token], W[label])  — one block per (token, model)
// ---------------------------------------------------------------------------
__global__ __launch_bounds__(256)
void dpo_tgt(const float* __restrict__ x, const float* __restrict__ rx,
             const int* __restrict__ y,
             const float* __restrict__ W, const float* __restrict__ rW,
             float* __restrict__ tgt) {
    const int token = blockIdx.x;
    const int model = blockIdx.y;
    const int tid = threadIdx.x;
    const int lab = y[token];
    if (lab == IGNORE_IDX) {
        if (tid == 0) tgt[model * TOKS + token] = 0.f;
        return;
    }
    const float* xr = (model ? rx : x) + (size_t)token * HDIM;
    const float* wr = (model ? rW : W) + (size_t)lab * HDIM;
    float s = 0.f;
    #pragma unroll
    for (int i = 0; i < 4; ++i) {
        const int o = (tid + i * 256) * 4;
        float4 a = *reinterpret_cast<const float4*>(xr + o);
        float4 b = *reinterpret_cast<const float4*>(wr + o);
        s += a.x * b.x + a.y * b.y + a.z * b.z + a.w * b.w;
    }
    #pragma unroll
    for (int off = 16; off >= 1; off >>= 1) s += __shfl_xor(s, off, 32);
    __shared__ float part[8];
    if ((tid & 31) == 0) part[tid >> 5] = s;
    __syncthreads();
    if (tid == 0) {
        float t = 0.f;
        #pragma unroll
        for (int j = 0; j < 8; ++j) t += part[j];
        tgt[model * TOKS + token] = t;
    }
}

// ---------------------------------------------------------------------------
// Kernel 3: combine per-chunk (max,sumexp) -> logZ -> token logprob
// partials layout [model][ntile][token][2]: reads coalesced across threads
// ---------------------------------------------------------------------------
__global__ void dpo_combine(const float* __restrict__ partials,
                            const float* __restrict__ tgt,
                            float* __restrict__ logp) {
    const int gid = blockIdx.x * blockDim.x + threadIdx.x;
    if (gid >= 2 * TOKS) return;
    const int model = gid >> 12;        // /4096
    const int token = gid & (TOKS - 1);
    float M = -3.0e38f, S = 0.f;
    for (int c = 0; c < NT; ++c) {
        const float* p = partials + (((size_t)model * NT + c) * TOKS + token) * 2;
        const float m = p[0], s = p[1];
        const float nM = fmaxf(M, m);
        S = S * __expf(M - nM) + s * __expf(m - nM);
        M = nM;
    }
    logp[gid] = tgt[gid] - (M + logf(S));
}

// ---------------------------------------------------------------------------
// Kernel 4: per-sequence masked averages + DPO loss (scalar)
// ---------------------------------------------------------------------------
__global__ void dpo_loss(const int* __restrict__ y,
                         const float* __restrict__ logp,
                         float* __restrict__ out) {
    __shared__ float avg[2][8];
    const int tid = threadIdx.x;
    if (tid < 16) {
        const int model = tid >> 3, b = tid & 7;
        float s = 0.f, c = 0.f;
        for (int t = 0; t < 512; ++t) {
            const int token = b * 512 + t;
            if (y[token] != IGNORE_IDX) { s += logp[model * TOKS + token]; c += 1.f; }
        }
        avg[model][b] = s / fmaxf(c, 1.f);
    }
    __syncthreads();
    if (tid == 0) {
        float loss = 0.f;
        #pragma unroll
        for (int i = 0; i < 4; ++i) {
            const float z = BETA * ((avg[0][i] - avg[1][i]) -
                                    (avg[0][i + 4] - avg[1][i + 4]));
            const float ls = fminf(z, 0.f) - log1pf(__expf(-fabsf(z)));
            loss += -ls;
        }
        out[0] = loss * 0.25f;
    }
}

// ---------------------------------------------------------------------------
extern "C" void kernel_launch(void* const* d_in, const int* in_sizes, int n_in,
                              void* d_out, int out_size, void* d_ws, size_t ws_size,
                              hipStream_t stream) {
    const float* x  = (const float*)d_in[0];
    const float* rx = (const float*)d_in[1];
    const int*   y  = (const int*)d_in[2];
    const float* W  = (const float*)d_in[3];
    const float* rW = (const float*)d_in[4];
    float* ws       = (float*)d_ws;

    float* partials = ws;                                   // 2*125*4096*2 = 2,048,000 f
    float* tgt      = ws + (size_t)2 * NT * TOKS * 2;       // 8192 f
    float* logp     = tgt + 2 * TOKS;                       // 8192 f
    float* out      = (float*)d_out;

    dim3 g1(NT, 2, 2);
    dpo_gemm_lse<<<g1, 256, 0, stream>>>(x, rx, W, rW, partials);

    dim3 g2(TOKS, 2);
    dpo_tgt<<<g2, 256, 0, stream>>>(x, rx, y, W, rW, tgt);

    dpo_combine<<<(2 * TOKS + 255) / 256, 256, 0, stream>>>(partials, tgt, logp);
    dpo_loss<<<1, 64, 0, stream>>>(y, logp, out);
}
